// VanillaRNN_34720515621547
// MI455X (gfx1250) — matrix operand order repaired and checked by reading, not compiled, not run on hardware
//
#include <hip/hip_runtime.h>
#include <hip/hip_bf16.h>

// ---------------------------------------------------------------------------
// VanillaRNN on MI455X (gfx1250): bf16 WMMA with fp32 accumulate.
//   B=128, T=1024, D_in=D_h=D_out=256
// ---------------------------------------------------------------------------

typedef __bf16 v16bf __attribute__((ext_vector_type(16)));
typedef float  v8f   __attribute__((ext_vector_type(8)));

struct BF16x16 { uint4 lo, hi; };             // 32 bytes == v16bf

static __device__ __forceinline__ unsigned short bf16_bits(float f) {
  return __builtin_bit_cast(unsigned short, (__bf16)f);
}

// A-fragment (16x32 bf16, MxK). Lane l&15 = row; ka already folds in the +8
// K-offset for lanes >= 16. p points at bf16 element [row][ka] (row-major).
static __device__ __forceinline__ v16bf load_a_frag_u16(const unsigned short* p) {
  BF16x16 t;
  t.lo = *(const uint4*)(p);        // K = ka .. ka+7
  t.hi = *(const uint4*)(p + 16);   // K = ka+16 .. ka+23
  return __builtin_bit_cast(v16bf, t);
}

// Build an A-fragment from an fp32 row (global), converting to bf16.
static __device__ __forceinline__ v16bf make_a_frag_f32(const float* rowp, int ka) {
  float4 a0 = *(const float4*)(rowp + ka);
  float4 a1 = *(const float4*)(rowp + ka + 4);
  float4 a2 = *(const float4*)(rowp + ka + 16);
  float4 a3 = *(const float4*)(rowp + ka + 20);
  v16bf r;
  r[0]  = (__bf16)a0.x; r[1]  = (__bf16)a0.y; r[2]  = (__bf16)a0.z; r[3]  = (__bf16)a0.w;
  r[4]  = (__bf16)a1.x; r[5]  = (__bf16)a1.y; r[6]  = (__bf16)a1.z; r[7]  = (__bf16)a1.w;
  r[8]  = (__bf16)a2.x; r[9]  = (__bf16)a2.y; r[10] = (__bf16)a2.z; r[11] = (__bf16)a2.w;
  r[12] = (__bf16)a3.x; r[13] = (__bf16)a3.y; r[14] = (__bf16)a3.z; r[15] = (__bf16)a3.w;
  return r;
}

// B-fragment (32x16 bf16, KxN) from a transposed weight WT[N][K]:
// lane holds column n, K = kb .. kb+7 and kb+16 .. kb+23.
static __device__ __forceinline__ v16bf load_b_frag(const unsigned short* WT,
                                                    int ncol, int kb) {
  BF16x16 t;
  t.lo = *(const uint4*)(WT + ncol * 256 + kb);
  t.hi = *(const uint4*)(WT + ncol * 256 + kb + 16);
  return __builtin_bit_cast(v16bf, t);
}

static __device__ __forceinline__ v8f wmma_bf16(v16bf a, v16bf b, v8f c) {
  return __builtin_amdgcn_wmma_f32_16x16x32_bf16(false, a, false, b, (short)0,
                                                 c, false, false);
}

// ---------------------------------------------------------------------------
// Kernel 1: transpose + convert the three 256x256 fp32 weights to bf16 [N][K].
// ---------------------------------------------------------------------------
__global__ __launch_bounds__(256) void rnn_convert_weights(
    const float* __restrict__ Wxh, const float* __restrict__ Whh,
    const float* __restrict__ Why,
    unsigned short* __restrict__ WxhT, unsigned short* __restrict__ WhhT,
    unsigned short* __restrict__ WhyT) {
  int idx = blockIdx.x * 256 + threadIdx.x;   // 65536 threads
  int n = idx >> 8;                            // output row  (column of W)
  int k = idx & 255;                           // output col  (row of W)
  int src = k * 256 + n;
  int dst = n * 256 + k;
  WxhT[dst] = bf16_bits(Wxh[src]);
  WhhT[dst] = bf16_bits(Whh[src]);
  WhyT[dst] = bf16_bits(Why[src]);
}

// ---------------------------------------------------------------------------
// Kernel 2: xW = x @ Wxh.  M = B*T = 131072, N = K = 256, fp32 out.
// Block = 256 threads (8 waves, 2M x 4N). Each wave: 16M x 64N tile.
// ---------------------------------------------------------------------------
__global__ __launch_bounds__(256) void rnn_xw_gemm(
    const float* __restrict__ x,              // [131072][256] fp32
    const unsigned short* __restrict__ WxhT,  // [256][256] bf16 (N-major)
    float* __restrict__ xW) {                 // [131072][256] fp32
  const int wave = threadIdx.x >> 5;
  const int lane = threadIdx.x & 31;
  const int waveM = wave >> 2;                // 0..1
  const int waveN = wave & 3;                 // 0..3
  const int m0 = blockIdx.x * 32 + waveM * 16;
  const int n0 = waveN * 64;
  const int lcol = lane & 15;
  const int half = lane >> 4;                 // 0 or 1

  v8f acc[4] = {v8f{}, v8f{}, v8f{}, v8f{}};

  const float* arow = x + (size_t)(m0 + lcol) * 256;

#pragma unroll
  for (int ks = 0; ks < 8; ++ks) {
    const int k0 = ks * 32;
    const int ka = k0 + (half << 3);          // +8 for lanes 16..31
    const int kb = k0 + (half << 3);          // +8 for lanes 16..31
    // Issue all loads for this k-step first, then the 4 WMMAs.
    v16bf bf[4];
#pragma unroll
    for (int j = 0; j < 4; ++j) bf[j] = load_b_frag(WxhT, n0 + j * 16 + lcol, kb);
    v16bf af = make_a_frag_f32(arow, ka);
#pragma unroll
    for (int j = 0; j < 4; ++j) acc[j] = wmma_bf16(af, bf[j], acc[j]);
  }

  const int rbase = half << 3;                // lanes>=16 hold rows 8..15
#pragma unroll
  for (int j = 0; j < 4; ++j) {
    const int col = n0 + j * 16 + lcol;
#pragma unroll
    for (int r = 0; r < 8; ++r) {
      xW[(size_t)(m0 + rbase + r) * 256 + col] = acc[j][r];
    }
  }
}

// ---------------------------------------------------------------------------
// Kernel 3: sequential recurrence + final projection.
// Grid = 8 blocks (16 batch rows each), block = 256 threads (8 waves).
// Whh B-fragments live in REGISTERS (128 VGPRs/lane, occupancy irrelevant at
// 8 blocks); h is double-buffered bf16 in LDS -> one barrier per timestep.
// Per step, per wave: 2 N-tiles x 8 K-steps of v_wmma_f32_16x16x32_bf16 fed
// only by ds_load_b128 (A) and resident registers (B).
// ---------------------------------------------------------------------------
__global__ __launch_bounds__(256) void rnn_scan(
    const float* __restrict__ xW,             // [128*1024][256] fp32
    const unsigned short* __restrict__ WhhT,  // [256][256] bf16
    const unsigned short* __restrict__ WhyT,  // [256][256] bf16
    const float* __restrict__ bh, const float* __restrict__ by,
    float* __restrict__ out) {                // [128][256] fp32
  __shared__ unsigned short hbuf[2][16 * 256];  // 2 x 8 KB: h as bf16, [m][k]

  const int wave = threadIdx.x >> 5;          // 0..7
  const int lane = threadIdx.x & 31;
  const int b0 = blockIdx.x * 16;
  const int n0 = wave * 32;
  const int lcol  = lane & 15;
  const int half  = lane >> 4;
  const int rbase = half << 3;

  // Preload ALL Whh B-fragments for this wave's 32 columns into registers.
  v16bf bw0[8], bw1[8];
#pragma unroll
  for (int ks = 0; ks < 8; ++ks) {
    const int kb = ks * 32 + (half << 3);
    bw0[ks] = load_b_frag(WhhT, n0 + lcol, kb);
    bw1[ks] = load_b_frag(WhhT, n0 + 16 + lcol, kb);
  }

  // h0 = 0 (read buffer 0)
  for (int i = threadIdx.x; i < 16 * 256; i += 256) hbuf[0][i] = 0;
  __syncthreads();

  const float bh0 = bh[n0 + lcol];
  const float bh1 = bh[n0 + 16 + lcol];

  int p = 0;
  for (int t = 0; t < 1024; ++t) {
    // Issue this step's xW loads up front; they are consumed only after the
    // WMMA chain, so their latency overlaps the matrix work.
    float xw0[8], xw1[8];
#pragma unroll
    for (int r = 0; r < 8; ++r) {
      const size_t row = (size_t)(b0 + rbase + r) * 1024 + t;
      xw0[r] = xW[row * 256 + n0 + lcol];
      xw1[r] = xW[row * 256 + n0 + 16 + lcol];
    }
    if (t < 1023) {
      __builtin_prefetch(
          xW + ((size_t)(b0 + rbase) * 1024 + (t + 1)) * 256 + n0 + lcol, 0, 1);
    }

    v8f acc0 = v8f{}, acc1 = v8f{};
#pragma unroll
    for (int ks = 0; ks < 8; ++ks) {
      const int ka = ks * 32 + (half << 3);
      v16bf af = load_a_frag_u16(&hbuf[p][lcol * 256 + ka]);
      acc0 = wmma_bf16(af, bw0[ks], acc0);
      acc1 = wmma_bf16(af, bw1[ks], acc1);
    }

    // Write new h into the OTHER buffer: no race with this step's readers.
    const int q = p ^ 1;
#pragma unroll
    for (int r = 0; r < 8; ++r) {
      float h0 = tanhf(xw0[r] + acc0[r] + bh0);
      float h1 = tanhf(xw1[r] + acc1[r] + bh1);
      hbuf[q][(rbase + r) * 256 + n0 + lcol]      = bf16_bits(h0);
      hbuf[q][(rbase + r) * 256 + n0 + 16 + lcol] = bf16_bits(h1);
    }
    __syncthreads();                          // publish h_{t+1} for all waves
    p = q;
  }

  // out = hT @ Why + by  (Why fragments loaded once from L2-hot global)
  v8f acc0 = v8f{}, acc1 = v8f{};
#pragma unroll
  for (int ks = 0; ks < 8; ++ks) {
    const int ka = ks * 32 + (half << 3);
    const int kb = ks * 32 + (half << 3);
    v16bf af = load_a_frag_u16(&hbuf[p][lcol * 256 + ka]);
    v16bf y0 = load_b_frag(WhyT, n0 + lcol, kb);
    v16bf y1 = load_b_frag(WhyT, n0 + 16 + lcol, kb);
    acc0 = wmma_bf16(af, y0, acc0);
    acc1 = wmma_bf16(af, y1, acc1);
  }
  const float by0 = by[n0 + lcol];
  const float by1 = by[n0 + 16 + lcol];
#pragma unroll
  for (int r = 0; r < 8; ++r) {
    out[(b0 + rbase + r) * 256 + n0 + lcol]      = acc0[r] + by0;
    out[(b0 + rbase + r) * 256 + n0 + 16 + lcol] = acc1[r] + by1;
  }
}

// ---------------------------------------------------------------------------
// Host launcher
// ---------------------------------------------------------------------------
extern "C" void kernel_launch(void* const* d_in, const int* in_sizes, int n_in,
                              void* d_out, int out_size, void* d_ws,
                              size_t ws_size, hipStream_t stream) {
  (void)in_sizes; (void)n_in; (void)out_size; (void)ws_size;
  const float* x   = (const float*)d_in[0];   // [128,1024,256]
  const float* Wxh = (const float*)d_in[1];   // [256,256]
  const float* Whh = (const float*)d_in[2];   // [256,256]
  const float* Why = (const float*)d_in[3];   // [256,256]
  const float* bh  = (const float*)d_in[4];   // [256]
  const float* by  = (const float*)d_in[5];   // [256]
  float* out = (float*)d_out;                 // [128,256]

  char* ws = (char*)d_ws;
  unsigned short* WxhT = (unsigned short*)(ws);                 // 128 KB
  unsigned short* WhhT = (unsigned short*)(ws + (128u << 10));  // 128 KB
  unsigned short* WhyT = (unsigned short*)(ws + (256u << 10));  // 128 KB
  float* xW = (float*)(ws + (512u << 10));                      // 128 MB

  rnn_convert_weights<<<256, 256, 0, stream>>>(Wxh, Whh, Why, WxhT, WhhT, WhyT);
  rnn_xw_gemm<<<4096, 256, 0, stream>>>(x, WxhT, xW);
  rnn_scan<<<8, 256, 0, stream>>>(xW, WhhT, WhyT, bh, by, out);
}
